// MultiHeadAttention_74723841016101
// MI455X (gfx1250) — compile-verified, matrix operand order checked by reading, not run on hardware
//
#include <hip/hip_runtime.h>

typedef __attribute__((ext_vector_type(16))) __bf16 v16bf;
typedef __attribute__((ext_vector_type(8)))  float  v8f;
typedef unsigned short u16;
typedef unsigned int   u32;

#define DIMB 8
#define SEQ  1024
#define DMODEL 1024
#define NHEAD 16
#define HD 64
#define BNROWS (DIMB*SEQ)   // 8192

// ---------- helpers ----------
__device__ __forceinline__ u16 f2bf(float x) {
    u32 u = __float_as_uint(x);
    u32 r = (u + 0x7FFFu + ((u >> 16) & 1u)) >> 16;   // round-to-nearest-even
    return (u16)r;
}

union FragU { uint4 u[2]; v16bf v; };
__device__ __forceinline__ v16bf make_frag(uint4 lo, uint4 hi) {
    FragU f; f.u[0] = lo; f.u[1] = hi; return f.v;
}

// CDNA5 async copy: global -> LDS, 16B per lane, tracked by ASYNCcnt.
// GVS addressing: 64-bit SGPR base + 32-bit per-lane byte offset.
__device__ __forceinline__ void async_b128(u32 lds_addr, u32 voff, const void* sbase) {
    asm volatile("global_load_async_to_lds_b128 %0, %1, %2"
                 :: "v"(lds_addr), "v"(voff), "s"(sbase) : "memory");
}
__device__ __forceinline__ void wait_async0() {
    asm volatile("s_wait_asynccnt 0x0" ::: "memory");
}
__device__ __forceinline__ u32 lds_off(const void* p) {   // flat -> LDS byte offset
    return (u32)(size_t)p;                                 // LDS offset = low 32 bits of the address
}

// ---------- fp32 -> bf16 weight transpose: wt[n][k] = w[k][n] ----------
__global__ __launch_bounds__(256) void wtrans_kernel(const float* __restrict__ w,
                                                     u16* __restrict__ wt) {
    int tid = blockIdx.x * 256 + threadIdx.x;          // 0 .. 1048575
    int n = tid >> 10, k = tid & 1023;
    wt[n * DMODEL + k] = f2bf(w[k * DMODEL + n]);
}

// ---------- fp32 -> bf16 activation convert (coalesced) ----------
__global__ __launch_bounds__(256) void xcvt_kernel(const float* __restrict__ x,
                                                   u16* __restrict__ xb) {
    int i = (blockIdx.x * 256 + threadIdx.x) * 4;
    float4 f = *(const float4*)(x + i);
    u32 lo = (u32)f2bf(f.x) | ((u32)f2bf(f.y) << 16);
    u32 hi = (u32)f2bf(f.z) | ((u32)f2bf(f.w) << 16);
    *(uint2*)(xb + i) = make_uint2(lo, hi);
}

// ---------- fused QKV projection GEMM (double-buffered async staging) ----------
// out = xb @ W + b ; xb bf16 [8192,1024], WT bf16 [1024 n][1024 k]
// z==0 -> Q [B,H,N,Hd], z==1 -> K [B,H,N,Hd], z==2 -> Vt [B,H,Hd,N]
#define LDT 40   // LDS row stride (bf16 elems), 80B, 16B-aligned rows

__global__ __launch_bounds__(256) void qkv_gemm_kernel(
    const u16* __restrict__ xb,
    const u16* __restrict__ WTq, const u16* __restrict__ WTk, const u16* __restrict__ WTv,
    const float* __restrict__ bq, const float* __restrict__ bk, const float* __restrict__ bv,
    u16* __restrict__ Qb, u16* __restrict__ Kb, u16* __restrict__ Vt)
{
    __shared__ __align__(16) u16 lA[2][128 * LDT];
    __shared__ __align__(16) u16 lB[2][128 * LDT];

    const int z = blockIdx.z;
    const u16*   WT   = (z == 0) ? WTq : (z == 1) ? WTk : WTv;
    const float* bias = (z == 0) ? bq  : (z == 1) ? bk  : bv;
    u16*         dst  = (z == 0) ? Qb  : (z == 1) ? Kb  : Vt;

    const int m0 = blockIdx.x * 128;
    const int n0 = blockIdx.y * 128;

    const int t    = threadIdx.x;
    const int wid  = t >> 5;
    const int lane = t & 31;
    const int lr   = lane & 15;
    const int lh   = lane >> 4;
    const int wm   = wid >> 1;        // 0..3 -> 32-row strip
    const int wn   = wid & 1;         // 0..1 -> 64-col strip

    // loader mapping: each thread stages 16 contiguous K elems (2 x b128) of one row
    const int lrow = t >> 1;
    const int lcol = (t & 1) * 16;
    const u32 voffA = (u32)(((m0 + lrow) * DMODEL + lcol) * 2);   // + kk*2 at issue
    const u32 voffB = (u32)(((n0 + lrow) * DMODEL + lcol) * 2);
    const u32 ldsAo = lrow * LDT + lcol;
    const u32 ldsBo = lrow * LDT + lcol;

    v8f acc[2][4];
#pragma unroll
    for (int i = 0; i < 2; ++i)
#pragma unroll
        for (int j = 0; j < 4; ++j) acc[i][j] = (v8f){};

    // ---- prologue: stage chunk 0 into buffer 0 ----
    {
        async_b128(lds_off(&lA[0][ldsAo]),     voffA,      xb);
        async_b128(lds_off(&lA[0][ldsAo + 8]), voffA + 16, xb);
        async_b128(lds_off(&lB[0][ldsBo]),     voffB,      WT);
        async_b128(lds_off(&lB[0][ldsBo + 8]), voffB + 16, WT);
    }
    wait_async0();
    __syncthreads();

    for (int ki = 0; ki < DMODEL / 32; ++ki) {
        const int cur = ki & 1;
        // ---- issue async stage of next chunk into the other buffer ----
        if (ki + 1 < DMODEL / 32) {
            const u32 koff = (u32)((ki + 1) * 32 * 2);
            async_b128(lds_off(&lA[cur ^ 1][ldsAo]),     voffA + koff,      xb);
            async_b128(lds_off(&lA[cur ^ 1][ldsAo + 8]), voffA + koff + 16, xb);
            async_b128(lds_off(&lB[cur ^ 1][ldsBo]),     voffB + koff,      WT);
            async_b128(lds_off(&lB[cur ^ 1][ldsBo + 8]), voffB + koff + 16, WT);
        }

        // ---- compute on current buffer ----
        v16bf afr[2], bfr[4];
#pragma unroll
        for (int tm = 0; tm < 2; ++tm) {
            int base = (wm * 32 + tm * 16 + lr) * LDT + lh * 8;
            afr[tm] = make_frag(*(const uint4*)&lA[cur][base],
                                *(const uint4*)&lA[cur][base + 16]);
        }
#pragma unroll
        for (int tn = 0; tn < 4; ++tn) {
            int base = (wn * 64 + tn * 16 + lr) * LDT + lh * 8;
            bfr[tn] = make_frag(*(const uint4*)&lB[cur][base],
                                *(const uint4*)&lB[cur][base + 16]);
        }
#pragma unroll
        for (int tm = 0; tm < 2; ++tm)
#pragma unroll
            for (int tn = 0; tn < 4; ++tn)
                acc[tm][tn] = __builtin_amdgcn_wmma_f32_16x16x32_bf16(
                    false, afr[tm], false, bfr[tn], (short)0, acc[tm][tn], false, false);

        wait_async0();       // next buffer staged
        __syncthreads();     // all waves done reading current buffer
    }

    // ---- epilogue: add bias, scatter to head-major layout ----
    float biasv[4];
#pragma unroll
    for (int tn = 0; tn < 4; ++tn)
        biasv[tn] = bias[n0 + wn * 64 + tn * 16 + lr];

#pragma unroll
    for (int tm = 0; tm < 2; ++tm) {
#pragma unroll
        for (int tn = 0; tn < 4; ++tn) {
            int gn = n0 + wn * 64 + tn * 16 + lr;
            int h = gn >> 6, d = gn & 63;
#pragma unroll
            for (int r = 0; r < 8; ++r) {
                int gm = m0 + wm * 32 + tm * 16 + lh * 8 + r;
                int bb = gm >> 10, nn = gm & 1023;
                float val = acc[tm][tn][r] + biasv[tn];
                if (z < 2)
                    dst[(((size_t)(bb * NHEAD + h) * SEQ) + nn) * HD + d] = f2bf(val);
                else
                    dst[(((size_t)(bb * NHEAD + h) * HD) + d) * SEQ + nn] = f2bf(val);
            }
        }
    }
}

// ---------- flash-style attention (double-buffered async K/V staging) ----------
// grid: (SEQ/128, B*H); block 256 = 8 waves; each wave owns 16 query rows.
#define LKS 72   // lds K row stride
#define LVS 40   // lds V row stride / lds P row stride

__global__ __launch_bounds__(256) void attn_kernel(
    const u16* __restrict__ Qb, const u16* __restrict__ Kb,
    const u16* __restrict__ Vt, float* __restrict__ out)
{
    __shared__ __align__(16) u16 lK[2][32 * LKS];      // [key][d]
    __shared__ __align__(16) u16 lV[2][64 * LVS];      // [d][key]
    __shared__ __align__(16) u16 lP[8 * 16 * LVS];     // per-wave P buffer [q][key]

    const int bh = blockIdx.y;
    const int b = bh >> 4, h = bh & 15;
    const int t = threadIdx.x;
    const int wid = t >> 5, lane = t & 31;
    const int lr = lane & 15, lh = lane >> 4;
    const int q0 = blockIdx.x * 128 + wid * 16;

    const size_t headoff = (size_t)(b * NHEAD + h) * SEQ * HD;  // == (b*16+h)*65536
    const u16* Qh = Qb + headoff;
    const u16* Kh = Kb + headoff;
    const u16* Vh = Vt + headoff;

    // per-thread staging addresses
    const int kr = t >> 3, kc = (t & 7) * 8;           // K chunk: 32 rows x 8 segs
    const int vr = t >> 2, vc = (t & 3) * 8;           // V chunk: 64 rows x 4 segs
    const u32 voffK = (u32)((kr * HD + kc) * 2);       // + kb*HD*2 at issue
    const u32 voffV = (u32)((vr * SEQ + vc) * 2);      // + kb*2 at issue
    const u32 ldsKo = kr * LKS + kc;
    const u32 ldsVo = vr * LVS + vc;

    // Q fragments for this wave's 16 rows (d = 0..31 and 32..63)
    v16bf qf[2];
#pragma unroll
    for (int f = 0; f < 2; ++f) {
        const u16* src = Qh + (size_t)(q0 + lr) * HD + f * 32 + lh * 8;
        qf[f] = make_frag(*(const uint4*)(src), *(const uint4*)(src + 16));
    }

    float mrow[8], lrow[8];
#pragma unroll
    for (int r = 0; r < 8; ++r) { mrow[r] = -1e30f; lrow[r] = 0.0f; }
    v8f o[4];
#pragma unroll
    for (int td = 0; td < 4; ++td) o[td] = (v8f){};

    u16* lPw = &lP[wid * 16 * LVS];

    // ---- prologue: stage key-chunk 0 into buffer 0 ----
    async_b128(lds_off(&lK[0][ldsKo]), voffK, Kh);
    async_b128(lds_off(&lV[0][ldsVo]), voffV, Vh);
    wait_async0();
    __syncthreads();

    for (int ki = 0; ki < SEQ / 32; ++ki) {
        const int cur = ki & 1;
        // ---- issue async stage of next key chunk ----
        if (ki + 1 < SEQ / 32) {
            async_b128(lds_off(&lK[cur ^ 1][ldsKo]), voffK + (u32)((ki + 1) * 32 * HD * 2), Kh);
            async_b128(lds_off(&lV[cur ^ 1][ldsVo]), voffV + (u32)((ki + 1) * 32 * 2),      Vh);
        }

        // ---- S = Q @ K^T (16 q x 32 keys), fp32 accum ----
        v8f s[2];
#pragma unroll
        for (int ct = 0; ct < 2; ++ct) {
            s[ct] = (v8f){};
#pragma unroll
            for (int f = 0; f < 2; ++f) {
                int base = (ct * 16 + lr) * LKS + f * 32 + lh * 8;
                v16bf kf = make_frag(*(const uint4*)&lK[cur][base],
                                     *(const uint4*)&lK[cur][base + 16]);
                s[ct] = __builtin_amdgcn_wmma_f32_16x16x32_bf16(
                    false, qf[f], false, kf, (short)0, s[ct], false, false);
            }
#pragma unroll
            for (int r = 0; r < 8; ++r) s[ct][r] *= 0.125f;        // 1/sqrt(64)
        }

        // ---- online softmax over this 32-key chunk ----
        float fscale[8];
#pragma unroll
        for (int r = 0; r < 8; ++r) {
            float v = fmaxf(s[0][r], s[1][r]);
#pragma unroll
            for (int off = 1; off < 16; off <<= 1)
                v = fmaxf(v, __shfl_xor(v, off, 32));              // 16-lane row reduce
            float mnew = fmaxf(mrow[r], v);
            float f = __expf(mrow[r] - mnew);
            mrow[r] = mnew;
            float p0 = __expf(s[0][r] - mnew);
            float p1 = __expf(s[1][r] - mnew);
            float rs = p0 + p1;
#pragma unroll
            for (int off = 1; off < 16; off <<= 1)
                rs += __shfl_xor(rs, off, 32);
            lrow[r] = lrow[r] * f + rs;
            s[0][r] = p0; s[1][r] = p1;
            fscale[r] = f;
        }
#pragma unroll
        for (int td = 0; td < 4; ++td)
#pragma unroll
            for (int r = 0; r < 8; ++r) o[td][r] *= fscale[r];

        // ---- C-layout P -> LDS -> A-fragment layout (wave-private, DS in-order) ----
#pragma unroll
        for (int ct = 0; ct < 2; ++ct)
#pragma unroll
            for (int r = 0; r < 8; ++r)
                lPw[(lh * 8 + r) * LVS + ct * 16 + lr] = f2bf(s[ct][r]);
        int pbase = lr * LVS + lh * 8;
        v16bf pf = make_frag(*(const uint4*)&lPw[pbase], *(const uint4*)&lPw[pbase + 16]);

        // ---- O += P @ V ----
#pragma unroll
        for (int td = 0; td < 4; ++td) {
            int base = (td * 16 + lr) * LVS + lh * 8;
            v16bf vf = make_frag(*(const uint4*)&lV[cur][base],
                                 *(const uint4*)&lV[cur][base + 16]);
            o[td] = __builtin_amdgcn_wmma_f32_16x16x32_bf16(
                false, pf, false, vf, (short)0, o[td], false, false);
        }

        wait_async0();       // next buffer staged
        __syncthreads();     // all waves done with current buffer
    }

    // ---- normalize + store fp32 [B, N, H*Hd] ----
    float inv[8];
#pragma unroll
    for (int r = 0; r < 8; ++r) inv[r] = 1.0f / (lrow[r] + 1e-30f);
#pragma unroll
    for (int td = 0; td < 4; ++td) {
#pragma unroll
        for (int r = 0; r < 8; ++r) {
            int gq = q0 + lh * 8 + r;
            int col = h * HD + td * 16 + lr;
            out[((size_t)(b * SEQ + gq)) * DMODEL + col] = o[td][r] * inv[r];
        }
    }
}

// ---------- host launcher ----------
extern "C" void kernel_launch(void* const* d_in, const int* in_sizes, int n_in,
                              void* d_out, int out_size, void* d_ws, size_t ws_size,
                              hipStream_t stream) {
    const float* x  = (const float*)d_in[0];
    const float* Wq = (const float*)d_in[1];
    const float* bq = (const float*)d_in[2];
    const float* Wk = (const float*)d_in[3];
    const float* bk = (const float*)d_in[4];
    const float* Wv = (const float*)d_in[5];
    const float* bv = (const float*)d_in[6];
    float* out = (float*)d_out;

    u16* ws = (u16*)d_ws;
    const size_t WSZ = (size_t)DMODEL * DMODEL;          // 1M elems per weight
    const size_t XSZ = (size_t)BNROWS * DMODEL;          // 8M elems
    const size_t QSZ = (size_t)DIMB * NHEAD * SEQ * HD;  // 8M elems
    u16* WTq = ws;
    u16* WTk = WTq + WSZ;
    u16* WTv = WTk + WSZ;
    u16* xb  = WTv + WSZ;
    u16* Qb  = xb + XSZ;
    u16* Kb  = Qb + QSZ;
    u16* Vt  = Kb + QSZ;
    (void)in_sizes; (void)n_in; (void)out_size; (void)ws_size;

    // 1) convert weights (transposed) and activations to bf16
    wtrans_kernel<<<(DMODEL * DMODEL) / 256, 256, 0, stream>>>(Wq, WTq);
    wtrans_kernel<<<(DMODEL * DMODEL) / 256, 256, 0, stream>>>(Wk, WTk);
    wtrans_kernel<<<(DMODEL * DMODEL) / 256, 256, 0, stream>>>(Wv, WTv);
    xcvt_kernel<<<(BNROWS * DMODEL) / (256 * 4), 256, 0, stream>>>(x, xb);

    // 2) fused QKV projections (z selects Q/K/V)
    dim3 ggrid(BNROWS / 128, DMODEL / 128, 3);
    qkv_gemm_kernel<<<ggrid, 256, 0, stream>>>(xb, WTq, WTk, WTv, bq, bk, bv, Qb, Kb, Vt);

    // 3) flash attention
    dim3 agrid(SEQ / 128, DIMB * NHEAD);
    attn_kernel<<<agrid, 256, 0, stream>>>(Qb, Kb, Vt, out);
}
